// ASTGCNBlock_5153960755472
// MI455X (gfx1250) — matrix operand order repaired and checked by reading, not compiled, not run on hardware
//
#include <hip/hip_runtime.h>

// ---------------- problem constants ----------------
#define Bq   8
#define Nq   1024
#define Fq   32
#define Tq   24
#define Eq   16384
#define Cq   64          // C_CHEB == C_TIME == 64
#define T2q  26          // T + 2 (zero padded for 1x3 temporal conv)
#define Mq   (Nq*Tq)     // 24576 rows per batch for cheb/time GEMMs

typedef __attribute__((ext_vector_type(16))) __bf16    v16bf;
typedef __attribute__((ext_vector_type(8)))  float     v8f;

__device__ __forceinline__ __bf16 us_as_bf(unsigned short s){
  union { unsigned short u; __bf16 b; } x; x.u = s; return x.b;
}
__device__ __forceinline__ unsigned short bf_as_us(__bf16 b){
  union { unsigned short u; __bf16 b; } x; x.b = b; return x.u;
}
__device__ __forceinline__ unsigned int pack2(float a, float b){
  return (unsigned int)bf_as_us((__bf16)a) | ((unsigned int)bf_as_us((__bf16)b) << 16);
}
__device__ __forceinline__ v8f vzero8(){
  v8f v;
#pragma unroll
  for (int i = 0; i < 8; ++i) v[i] = 0.f;
  return v;
}

// A fragment: 16x32 bf16 slice at K-offset koff of a [16 x lda] LDS tile.
// lanes 0-15 rows M=lane (K koff+0..7,16..23), lanes 16-31 M=lane-16 (K koff+8..15,24..31).
__device__ __forceinline__ v16bf load_A_frag(const __bf16* As, int lda, int koff){
  int lane = threadIdx.x & 31;
  int m  = lane & 15;
  int kb = koff + (lane >> 4) * 8;
  v16bf a;
#pragma unroll
  for (int e = 0; e < 16; ++e){
    int k = kb + e + ((e >= 8) ? 8 : 0);
    a[e] = As[m * lda + k];
  }
  return a;
}
// B fragment from LDS staged transposed BsT[n][k] (ld = ldb), K-offset koff:
// lanes 0-15 col n=lane (K koff+0..7,16..23), lanes 16-31 col n=lane-16 (K koff+8..15,24..31).
__device__ __forceinline__ v16bf load_B_frag(const __bf16* BsT, int ldb, int ncol_off, int koff){
  int lane = threadIdx.x & 31;
  int n  = (lane & 15) + ncol_off;
  int kb = koff + (lane >> 4) * 8;
  v16bf b;
#pragma unroll
  for (int e = 0; e < 16; ++e) b[e] = BsT[n * ldb + kb + e + ((e >= 8) ? 8 : 0)];
  return b;
}

// ---------------- elementwise / small kernels ----------------
__global__ void k_zero(float* p, size_t n){
  size_t i = (size_t)blockIdx.x * blockDim.x + threadIdx.x;
  if (i < n) p[i] = 0.f;
}

// tempTF[b,t,f] = sum_n X[b,n,f,t] * U1[n]
__global__ void k_tempTF(const float* __restrict__ X, const float* __restrict__ U1,
                         float* __restrict__ tmp){
  int id = blockIdx.x * blockDim.x + threadIdx.x;
  if (id >= Bq*Tq*Fq) return;
  int f = id % Fq, t = (id / Fq) % Tq, b = id / (Fq*Tq);
  float s = 0.f;
  for (int n = 0; n < Nq; ++n)
    s += X[(((size_t)b*Nq + n)*Fq + f)*Tq + t] * U1[n];
  tmp[id] = s;
}
// rhs[b,n,t] = sum_f U3[f] * X[b,n,f,t]
__global__ void k_rhs(const float* __restrict__ X, const float* __restrict__ U3,
                      float* __restrict__ rhs){
  int id = blockIdx.x * blockDim.x + threadIdx.x;
  if (id >= Bq*Nq*Tq) return;
  int t = id % Tq, n = (id / Tq) % Nq, b = id / (Tq*Nq);
  const float* xp = X + (((size_t)b*Nq + n)*Fq)*Tq + t;
  float s = 0.f;
  for (int f = 0; f < Fq; ++f) s += U3[f] * xp[(size_t)f*Tq];
  rhs[id] = s;
}
// lhs[b,t,n] = sum_f tempTF[b,t,f] * U2[f,n]
__global__ void k_lhs(const float* __restrict__ tmp, const float* __restrict__ U2,
                      float* __restrict__ lhs){
  int id = blockIdx.x * blockDim.x + threadIdx.x;
  if (id >= Bq*Tq*Nq) return;
  int n = id % Nq, t = (id / Nq) % Tq, b = id / (Nq*Tq);
  const float* tp = tmp + ((size_t)b*Tq + t)*Fq;
  float s = 0.f;
  for (int f = 0; f < Fq; ++f) s += tp[f] * U2[(size_t)f*Nq + n];
  lhs[id] = s;
}
// P[b,i,j] = sigmoid( sum_n lhs[b,i,n]*rhs[b,n,j] + be[i,j] )
__global__ void k_P(const float* __restrict__ lhs, const float* __restrict__ rhs,
                    const float* __restrict__ be, float* __restrict__ P){
  int id = blockIdx.x * blockDim.x + threadIdx.x;
  if (id >= Bq*Tq*Tq) return;
  int j = id % Tq, i = (id / Tq) % Tq, b = id / (Tq*Tq);
  const float* lp = lhs + ((size_t)b*Tq + i)*Nq;
  const float* rp = rhs + (size_t)b*Nq*Tq + j;
  float s = 0.f;
  for (int n = 0; n < Nq; ++n) s += lp[n] * rp[(size_t)n*Tq];
  s += be[i*Tq + j];
  P[id] = 1.f / (1.f + __expf(-s));
}
// Escore = Ve @ P ; softmax over axis=1 (row index i) per (b,j)
__global__ void k_Eatt(const float* __restrict__ Ve, const float* __restrict__ P,
                       float* __restrict__ Eatt){
  int b = blockIdx.x / Tq, j = blockIdx.x % Tq;
  int i = threadIdx.x;
  __shared__ float e[Tq];
  float v = 0.f;
  if (i < Tq){
    for (int k = 0; k < Tq; ++k) v += Ve[i*Tq + k] * P[((size_t)b*Tq + k)*Tq + j];
    e[i] = v;
  }
  __syncthreads();
  if (i < Tq){
    float m = -1e30f;
    for (int k = 0; k < Tq; ++k) m = fmaxf(m, e[k]);
    float s = 0.f;
    for (int k = 0; k < Tq; ++k) s += __expf(e[k] - m);
    Eatt[((size_t)b*Tq + i)*Tq + j] = __expf(v - m) / s;
  }
}
// Xt[b,nf,t'] = sum_t X[b,nf,t] * Eatt[b,t,t']
__global__ void k_Xt(const float* __restrict__ X, const float* __restrict__ Eatt,
                     float* __restrict__ Xt){
  __shared__ float Ea[Tq*Tq];
  int b   = blockIdx.x >> 7;           // 128 blocks per batch
  int row = (blockIdx.x & 127) * 256 + threadIdx.x;  // nf in [0, N*F)
  for (int e = threadIdx.x; e < Tq*Tq; e += 256) Ea[e] = Eatt[(size_t)b*Tq*Tq + e];
  __syncthreads();
  const float* xp = X  + ((size_t)b*Nq*Fq + row)*Tq;
  float*       op = Xt + ((size_t)b*Nq*Fq + row)*Tq;
  float xv[Tq];
#pragma unroll
  for (int t = 0; t < Tq; ++t) xv[t] = xp[t];
#pragma unroll
  for (int tp = 0; tp < Tq; ++tp){
    float s = 0.f;
#pragma unroll
    for (int t = 0; t < Tq; ++t) s += xv[t] * Ea[t*Tq + tp];
    op[tp] = s;
  }
}
// slhs0[b,n,f] = sum_t Xt * W1[t]
__global__ void k_slhs0(const float* __restrict__ Xt, const float* __restrict__ W1,
                        float* __restrict__ o){
  int id = blockIdx.x * blockDim.x + threadIdx.x;
  if (id >= Bq*Nq*Fq) return;
  const float* xp = Xt + (size_t)id * Tq;
  float s = 0.f;
  for (int t = 0; t < Tq; ++t) s += xp[t] * W1[t];
  o[id] = s;
}
// slhs[b,n,t] = sum_f slhs0[b,n,f] * W2[f,t]
__global__ void k_slhs(const float* __restrict__ s0, const float* __restrict__ W2,
                       float* __restrict__ o){
  int id = blockIdx.x * blockDim.x + threadIdx.x;
  if (id >= Bq*Nq*Tq) return;
  int t = id % Tq; size_t bn = id / Tq;
  const float* sp = s0 + bn*Fq;
  float s = 0.f;
  for (int f = 0; f < Fq; ++f) s += sp[f] * W2[f*Tq + t];
  o[id] = s;
}
// srhs[b,t,n] = sum_f W3[f] * Xt[b,n,f,t]
__global__ void k_srhs(const float* __restrict__ Xt, const float* __restrict__ W3,
                       float* __restrict__ o){
  int id = blockIdx.x * blockDim.x + threadIdx.x;
  if (id >= Bq*Nq*Tq) return;
  int t = id % Tq, n = (id / Tq) % Nq, b = id / (Tq*Nq);
  const float* xp = Xt + (((size_t)b*Nq + n)*Fq)*Tq + t;
  float s = 0.f;
  for (int f = 0; f < Fq; ++f) s += W3[f] * xp[(size_t)f*Tq];
  o[((size_t)b*Tq + t)*Nq + n] = s;
}
// sigP[b,i,j] = sigmoid(slhs[b,i,:]·srhs[b,:,j] + bs[i,j]) -> bf16
__global__ void k_sigP(const float* __restrict__ slhs, const float* __restrict__ srhs,
                       const float* __restrict__ bs, unsigned short* __restrict__ o){
  size_t id = (size_t)blockIdx.x * blockDim.x + threadIdx.x;
  if (id >= (size_t)Bq*Nq*Nq) return;
  int j = id % Nq; int i = (id / Nq) % Nq; int b = id / ((size_t)Nq*Nq);
  const float* lp = slhs + ((size_t)b*Nq + i)*Tq;
  const float* rp = srhs + (size_t)b*Tq*Nq + j;
  float s = 0.f;
#pragma unroll
  for (int t = 0; t < Tq; ++t) s += lp[t] * rp[(size_t)t*Nq];
  s += bs[(size_t)i*Nq + j];
  s = 1.f / (1.f + __expf(-s));
  o[id] = bf_as_us((__bf16)s);
}
__global__ void k_cvt_bf(const float* __restrict__ s, unsigned short* __restrict__ d, size_t n){
  size_t i = (size_t)blockIdx.x * blockDim.x + threadIdx.x;
  if (i < n) d[i] = bf_as_us((__bf16)s[i]);
}
// pack time_W [Cout,Cin,1,3] -> Wt2[k = j*64+cin][cout]  (bf16, 192x64)
__global__ void k_packWt(const float* __restrict__ W, unsigned short* __restrict__ o){
  int id = blockIdx.x * blockDim.x + threadIdx.x;
  if (id >= 192*64) return;
  int cout = id % 64, kk = id / 64;
  int j = kk / 64, cin = kk % 64;
  o[kk*64 + cout] = bf_as_us((__bf16)W[((size_t)cout*64 + cin)*3 + j]);
}

// ---------------- WMMA GEMM: Sscore = Vs(bf16) @ sigP(bf16), M=N=K=1024 ----------------
// 128 threads (4 wave32), 64x64 C-tile, K-step 64, LDS double tiles 16 KB.
__launch_bounds__(128)
__global__ void k_gemm_S(const unsigned short* __restrict__ Abf,
                         const unsigned short* __restrict__ Bbf,
                         float* __restrict__ C){
  __shared__ __align__(16) __bf16 As[64*64];   // [m][k] ld 64
  __shared__ __align__(16) __bf16 Bs[64*64];   // [n][k] ld 64 (transposed)
  int b = blockIdx.z;
  int row0 = blockIdx.y * 64, col0 = blockIdx.x * 64;
  int tid = threadIdx.x, wave = tid >> 5, lane = tid & 31;
  const unsigned short* Bg = Bbf + (size_t)b*Nq*Nq;
  v8f acc[4];
#pragma unroll
  for (int j = 0; j < 4; ++j) acc[j] = vzero8();
  for (int k0 = 0; k0 < Nq; k0 += 64){
    // prefetch next K-tile rows (global_prefetch_b8)
    if (k0 + 64 < Nq){
      int pr = tid & 63;
      __builtin_prefetch(Abf + (size_t)(row0 + pr)*Nq + k0 + 64, 0, 1);
      __builtin_prefetch(Bg  + (size_t)(k0 + 64 + pr)*Nq + col0, 0, 1);
    }
    // A tile: 4096 bf16, vector b128 loads + stores
#pragma unroll
    for (int p = 0; p < 4; ++p){
      int idx = (p*128 + tid) * 8;
      int r = idx >> 6, c = idx & 63;
      uint4 w = *(const uint4*)(Abf + (size_t)(row0 + r)*Nq + k0 + c);
      *(uint4*)(&As[idx]) = w;
    }
    // B tile: b128 global loads, transposed scalar LDS stores
#pragma unroll
    for (int p = 0; p < 4; ++p){
      int idx = (p*128 + tid) * 8;
      int k = idx >> 6, n = idx & 63;
      uint4 w = *(const uint4*)(Bg + (size_t)(k0 + k)*Nq + col0 + n);
      const unsigned short* wsv = (const unsigned short*)&w;
#pragma unroll
      for (int q = 0; q < 8; ++q) Bs[(n + q)*64 + k] = us_as_bf(wsv[q]);
    }
    __syncthreads();
#pragma unroll
    for (int h = 0; h < 2; ++h){
      v16bf a = load_A_frag(As + wave*16*64, 64, h*32);
#pragma unroll
      for (int j = 0; j < 4; ++j){
        v16bf bb = load_B_frag(Bs, 64, j*16, h*32);
        acc[j] = __builtin_amdgcn_wmma_f32_16x16x32_bf16(false, a, false, bb,
                                                         (short)0, acc[j], false, false);
      }
    }
    __syncthreads();
  }
  float* Cb = C + (size_t)b*Nq*Nq;
  int mb = row0 + wave*16 + 8*(lane >> 4);
  int nl = lane & 15;
#pragma unroll
  for (int j = 0; j < 4; ++j)
#pragma unroll
    for (int r = 0; r < 8; ++r)
      Cb[(size_t)(mb + r)*Nq + col0 + j*16 + nl] = acc[j][r];
}

// softmax over axis=1 (rows) of S[b,:,:] in place; grid(32, B), block(32,8)
__global__ void k_softmaxS(float* __restrict__ S){
  int b = blockIdx.y;
  int j = blockIdx.x*32 + threadIdx.x;
  float* Sb = S + (size_t)b*Nq*Nq;
  __shared__ float red[8][33];
  float m = -1e30f;
  for (int i = threadIdx.y; i < Nq; i += 8) m = fmaxf(m, Sb[(size_t)i*Nq + j]);
  red[threadIdx.y][threadIdx.x] = m; __syncthreads();
  if (threadIdx.y == 0){
    float mm = red[0][threadIdx.x];
    for (int y = 1; y < 8; ++y) mm = fmaxf(mm, red[y][threadIdx.x]);
    red[0][threadIdx.x] = mm;
  }
  __syncthreads();
  m = red[0][threadIdx.x];
  __syncthreads();
  float s = 0.f;
  for (int i = threadIdx.y; i < Nq; i += 8) s += __expf(Sb[(size_t)i*Nq + j] - m);
  red[threadIdx.y][threadIdx.x] = s; __syncthreads();
  if (threadIdx.y == 0){
    float ss = red[0][threadIdx.x];
    for (int y = 1; y < 8; ++y) ss += red[y][threadIdx.x];
    red[0][threadIdx.x] = ss;
  }
  __syncthreads();
  float inv = 1.f / red[0][threadIdx.x];
  for (int i = threadIdx.y; i < Nq; i += 8){
    size_t o = (size_t)i*Nq + j;
    Sb[o] = __expf(Sb[o] - m) * inv;
  }
}
__global__ void k_diag(const float* __restrict__ S, float* __restrict__ d){
  int id = blockIdx.x * blockDim.x + threadIdx.x;
  if (id >= Bq*Nq) return;
  int n = id % Nq, b = id / Nq;
  d[id] = S[(size_t)b*Nq*Nq + (size_t)n*Nq + n];
}

// ---------------- graph Laplacian pieces ----------------
__global__ void k_deg(const int* __restrict__ ei, float* __restrict__ deg){
  int e = blockIdx.x * blockDim.x + threadIdx.x;
  if (e >= Eq) return;
  int r = ei[e], c = ei[Eq + e];
  if (r != c) atomicAdd(&deg[r], 1.f);
}
__global__ void k_dis(const float* __restrict__ deg, float* __restrict__ dis){
  int n = blockIdx.x * blockDim.x + threadIdx.x;
  if (n >= Nq) return;
  float d = deg[n];
  dis[n] = (d > 0.f) ? rsqrtf(d) : 0.f;
}
__global__ void k_norm(const int* __restrict__ ei, const float* __restrict__ dis,
                       float* __restrict__ nrm){
  int e = blockIdx.x * blockDim.x + threadIdx.x;
  if (e >= Eq) return;
  int r = ei[e], c = ei[Eq + e];
  float w = (r != c) ? 1.f : 0.f;
  nrm[e] = -dis[r] * w * dis[c];
}
__global__ void k_attnorm(const float* __restrict__ nrm, const float* __restrict__ S,
                          const int* __restrict__ ei, float* __restrict__ an){
  int id = blockIdx.x * blockDim.x + threadIdx.x;
  if (id >= Bq*Eq) return;
  int e = id % Eq, b = id / Eq;
  an[id] = nrm[e] * S[(size_t)b*Nq*Nq + (size_t)ei[e]*Nq + ei[Eq + e]];
}
// Tx0L[b,n,t,f] = diagS[b,n] * X[b,n,f,t]
__global__ void k_Tx0(const float* __restrict__ X, const float* __restrict__ dg,
                      float* __restrict__ o){
  size_t id = (size_t)blockIdx.x * blockDim.x + threadIdx.x;
  if (id >= (size_t)Bq*Nq*Tq*Fq) return;
  int f = id % Fq; int t = (id / Fq) % Tq;
  size_t bn = id / ((size_t)Fq*Tq);
  o[id] = dg[bn] * X[(bn*Fq + f)*Tq + t];
}
// dst[b,row,:,:] += scale*coef[b?,e] * src[b,col,:,:]  over edges; grid(E, B)
__global__ void k_prop(float* __restrict__ dst, const float* __restrict__ src,
                       const float* __restrict__ coefs, int strideB, float scale,
                       const int* __restrict__ ei){
  int e = blockIdx.x, b = blockIdx.y;
  float coef = coefs[(size_t)b*strideB + e] * scale;
  int r = ei[e], c = ei[Eq + e];
  const float* sp = src + ((size_t)b*Nq + c)*(Tq*Fq);
  float*       dp = dst + ((size_t)b*Nq + r)*(Tq*Fq);
  for (int i = threadIdx.x; i < Tq*Fq; i += 256)
    atomicAdd(dp + i, coef * sp[i]);
}
__global__ void k_negcopy(const float* __restrict__ s, float* __restrict__ d, size_t n){
  size_t i = (size_t)blockIdx.x * blockDim.x + threadIdx.x;
  if (i < n) d[i] = -s[i];
}

// ---------------- WMMA GEMM: cheb  acc[b, n, t+1, c] (+)= TxL[b,(n,t),f] @ W[f,c] ----------------
__launch_bounds__(128)
__global__ void k_gemm_cheb(const float* __restrict__ A, const unsigned short* __restrict__ Wbf,
                            float* __restrict__ acc, int accumulate){
  __shared__ __align__(16) __bf16 As[64*32];
  __shared__ __align__(16) __bf16 Bs[64*32];   // [c][f]
  int b = blockIdx.y, m0 = blockIdx.x * 64;
  int tid = threadIdx.x, wave = tid >> 5, lane = tid & 31;
  const float* Ab = A + (size_t)b*Mq*Fq;
  // A tile: 2048 f32 -> bf16, vectorized (2 x 8 per thread)
#pragma unroll
  for (int p = 0; p < 2; ++p){
    int idx = (p*128 + tid) * 8;
    int r = idx >> 5, c = idx & 31;
    const float* src = Ab + (size_t)(m0 + r)*Fq + c;
    float4 f0 = *(const float4*)(src);
    float4 f1 = *(const float4*)(src + 4);
    uint4 q;
    q.x = pack2(f0.x, f0.y); q.y = pack2(f0.z, f0.w);
    q.z = pack2(f1.x, f1.y); q.w = pack2(f1.z, f1.w);
    *(uint4*)(&As[idx]) = q;
  }
  // B tile (32x64 bf16) -> transposed [c][f]
#pragma unroll
  for (int p = 0; p < 2; ++p){
    int idx = (p*128 + tid) * 8;
    int f = idx >> 6, c = idx & 63;
    uint4 w = *(const uint4*)(Wbf + (size_t)f*64 + c);
    const unsigned short* wsv = (const unsigned short*)&w;
#pragma unroll
    for (int q = 0; q < 8; ++q) Bs[(c + q)*32 + f] = us_as_bf(wsv[q]);
  }
  __syncthreads();
  v16bf a = load_A_frag(As + wave*16*32, 32, 0);
  v8f av[4];
#pragma unroll
  for (int j = 0; j < 4; ++j){
    av[j] = vzero8();
    v16bf bb = load_B_frag(Bs, 32, j*16, 0);
    av[j] = __builtin_amdgcn_wmma_f32_16x16x32_bf16(false, a, false, bb,
                                                    (short)0, av[j], false, false);
  }
  float* ob = acc + (size_t)b*Nq*T2q*Cq;
  int mb = m0 + wave*16 + 8*(lane >> 4);
  int nl = lane & 15;
#pragma unroll
  for (int j = 0; j < 4; ++j)
#pragma unroll
    for (int r = 0; r < 8; ++r){
      int m = mb + r, n = m / Tq, t = m % Tq;
      size_t o = ((size_t)n*T2q + t + 1)*Cq + j*16 + nl;
      if (accumulate) ob[o] += av[j][r]; else ob[o] = av[j][r];
    }
}
// relu(acc + cheb_b) on interior rows, zero on the two pad rows per n
__global__ void k_biasrelu(float* __restrict__ acc, const float* __restrict__ cb){
  size_t id = (size_t)blockIdx.x * blockDim.x + threadIdx.x;
  if (id >= (size_t)Bq*Nq*T2q*Cq) return;
  int c = id % Cq, t = (id / Cq) % T2q;
  if (t == 0 || t == T2q - 1) acc[id] = 0.f;
  else                        acc[id] = fmaxf(acc[id] + cb[c], 0.f);
}

// ---------------- WMMA GEMM: temporal conv im2col  (K = 192 contiguous, K-step 64) ----------------
__launch_bounds__(128)
__global__ void k_gemm_time(const float* __restrict__ Xp, const unsigned short* __restrict__ Wbf,
                            const float* __restrict__ tb, float* __restrict__ tout){
  __shared__ __align__(16) __bf16 As[64*64];
  __shared__ __align__(16) __bf16 Bs[64*64];
  int b = blockIdx.y, m0 = blockIdx.x * 64;
  int tid = threadIdx.x, wave = tid >> 5, lane = tid & 31;
  const float* Xb = Xp + (size_t)b*Nq*T2q*Cq;
  v8f acc[4];
#pragma unroll
  for (int j = 0; j < 4; ++j) acc[j] = vzero8();
  for (int k0 = 0; k0 < 192; k0 += 64){
    // A tile: 4096 f32 -> bf16, vectorized convert+pack
#pragma unroll
    for (int p = 0; p < 4; ++p){
      int idx = (p*128 + tid) * 8;
      int r = idx >> 6, c = idx & 63;
      int m = m0 + r, n = m / Tq, t = m % Tq;       // window starts at padded idx t
      const float* src = Xb + ((size_t)n*T2q + t)*Cq + k0 + c;
      float4 f0 = *(const float4*)(src);
      float4 f1 = *(const float4*)(src + 4);
      uint4 q;
      q.x = pack2(f0.x, f0.y); q.y = pack2(f0.z, f0.w);
      q.z = pack2(f1.x, f1.y); q.w = pack2(f1.z, f1.w);
      *(uint4*)(&As[idx]) = q;
    }
    // B tile: b128 global loads, transposed scalar LDS stores
#pragma unroll
    for (int p = 0; p < 4; ++p){
      int idx = (p*128 + tid) * 8;
      int k = idx >> 6, n = idx & 63;
      uint4 w = *(const uint4*)(Wbf + (size_t)(k0 + k)*64 + n);
      const unsigned short* wsv = (const unsigned short*)&w;
#pragma unroll
      for (int q = 0; q < 8; ++q) Bs[(n + q)*64 + k] = us_as_bf(wsv[q]);
    }
    __syncthreads();
#pragma unroll
    for (int h = 0; h < 2; ++h){
      v16bf a = load_A_frag(As + wave*16*64, 64, h*32);
#pragma unroll
      for (int j = 0; j < 4; ++j){
        v16bf bb = load_B_frag(Bs, 64, j*16, h*32);
        acc[j] = __builtin_amdgcn_wmma_f32_16x16x32_bf16(false, a, false, bb,
                                                         (short)0, acc[j], false, false);
      }
    }
    __syncthreads();
  }
  float* ob = tout + (size_t)b*Mq*Cq;
  int mb = m0 + wave*16 + 8*(lane >> 4);
  int nl = lane & 15;
#pragma unroll
  for (int j = 0; j < 4; ++j)
#pragma unroll
    for (int r = 0; r < 8; ++r){
      int col = j*16 + nl;
      ob[(size_t)(mb + r)*Cq + col] = acc[j][r] + tb[col];
    }
}

// ---------------- fused residual conv + relu + LayerNorm + output transpose ----------------
__global__ void k_final(const float* __restrict__ tout, const float* __restrict__ X,
                        const float* __restrict__ rW, const float* __restrict__ rb,
                        const float* __restrict__ lg, const float* __restrict__ lb,
                        float* __restrict__ out){
  int id = blockIdx.x;                 // b*N*T + n*T + t
  int t = id % Tq, n = (id / Tq) % Nq, b = id / (Tq*Nq);
  int c = threadIdx.x;                 // 64
  __shared__ float xsh[Fq];
  __shared__ float sv[Cq], sq[Cq];
  if (c < Fq) xsh[c] = X[(((size_t)b*Nq + n)*Fq + c)*Tq + t];
  __syncthreads();
  float r = rb[c];
#pragma unroll
  for (int f = 0; f < Fq; ++f) r += rW[c*Fq + f] * xsh[f];
  float v = fmaxf(tout[(size_t)id*Cq + c] + r, 0.f);
  sv[c] = v; sq[c] = v * v; __syncthreads();
  for (int s = 32; s > 0; s >>= 1){
    if (c < s){ sv[c] += sv[c + s]; sq[c] += sq[c + s]; }
    __syncthreads();
  }
  float mu  = sv[0] / Cq;
  float var = sq[0] / Cq - mu * mu;
  out[(((size_t)b*Nq + n)*Cq + c)*Tq + t] =
      (v - mu) * rsqrtf(var + 1e-5f) * lg[c] + lb[c];
}

// ---------------- launch ----------------
extern "C" void kernel_launch(void* const* d_in, const int* in_sizes, int n_in,
                              void* d_out, int out_size, void* d_ws, size_t ws_size,
                              hipStream_t stream){
  const float* X    = (const float*)d_in[0];
  const int*   ei   = (const int*)  d_in[1];
  const float* U1   = (const float*)d_in[2];
  const float* U2   = (const float*)d_in[3];
  const float* U3   = (const float*)d_in[4];
  const float* be   = (const float*)d_in[5];
  const float* Ve   = (const float*)d_in[6];
  const float* W1   = (const float*)d_in[7];
  const float* W2   = (const float*)d_in[8];
  const float* W3   = (const float*)d_in[9];
  const float* bs   = (const float*)d_in[10];
  const float* Vs   = (const float*)d_in[11];
  const float* chW  = (const float*)d_in[12];
  const float* chb  = (const float*)d_in[13];
  const float* tW   = (const float*)d_in[14];
  const float* tb   = (const float*)d_in[15];
  const float* rW   = (const float*)d_in[16];
  const float* rb   = (const float*)d_in[17];
  const float* lg   = (const float*)d_in[18];
  const float* lb   = (const float*)d_in[19];
  float* out = (float*)d_out;
  float* ws  = (float*)d_ws;

  // workspace layout (floats); ~263 MB total
  size_t o = 0;
  float* tempTF = ws + o; o += (size_t)Bq*Tq*Fq;
  float* rhs    = ws + o; o += (size_t)Bq*Nq*Tq;
  float* lhs    = ws + o; o += (size_t)Bq*Tq*Nq;
  float* P      = ws + o; o += (size_t)Bq*Tq*Tq;
  float* Eatt   = ws + o; o += (size_t)Bq*Tq*Tq;
  float* Xt     = ws + o; o += (size_t)Bq*Nq*Fq*Tq;
  float* slhs0  = ws + o; o += (size_t)Bq*Nq*Fq;
  float* slhs   = ws + o; o += (size_t)Bq*Nq*Tq;
  float* srhs   = ws + o; o += (size_t)Bq*Tq*Nq;
  unsigned short* Vsbf  = (unsigned short*)(ws + o); o += (size_t)Nq*Nq/2;
  unsigned short* sigPb = (unsigned short*)(ws + o); o += (size_t)Bq*Nq*Nq/2;
  float* S      = ws + o; o += (size_t)Bq*Nq*Nq;
  float* deg    = ws + o; o += Nq;
  float* dis    = ws + o; o += Nq;
  float* nrm    = ws + o; o += Eq;
  float* attn   = ws + o; o += (size_t)Bq*Eq;
  float* dgS    = ws + o; o += (size_t)Bq*Nq;
  float* Tx0    = ws + o; o += (size_t)Bq*Mq*Fq;
  float* Tx1    = ws + o; o += (size_t)Bq*Mq*Fq;
  float* Tx2    = ws + o; o += (size_t)Bq*Mq*Fq;
  unsigned short* chWbf = (unsigned short*)(ws + o); o += (size_t)3*Fq*Cq/2;
  unsigned short* Wt2   = (unsigned short*)(ws + o); o += (size_t)192*64/2;
  float* acc    = ws + o; o += (size_t)Bq*Nq*T2q*Cq;
  float* tout   = ws + o; o += (size_t)Bq*Mq*Cq;

  const int TB = 256;
  // temporal attention
  k_tempTF<<<dim3((Bq*Tq*Fq + TB-1)/TB), TB, 0, stream>>>(X, U1, tempTF);
  k_rhs   <<<dim3((Bq*Nq*Tq + TB-1)/TB), TB, 0, stream>>>(X, U3, rhs);
  k_lhs   <<<dim3((Bq*Tq*Nq + TB-1)/TB), TB, 0, stream>>>(tempTF, U2, lhs);
  k_P     <<<dim3((Bq*Tq*Tq + 63)/64), 64, 0, stream>>>(lhs, rhs, be, P);
  k_Eatt  <<<dim3(Bq*Tq), 32, 0, stream>>>(Ve, P, Eatt);
  k_Xt    <<<dim3(Bq*128), 256, 0, stream>>>(X, Eatt, Xt);
  // spatial attention
  k_slhs0 <<<dim3((Bq*Nq*Fq + TB-1)/TB), TB, 0, stream>>>(Xt, W1, slhs0);
  k_slhs  <<<dim3((Bq*Nq*Tq + TB-1)/TB), TB, 0, stream>>>(slhs0, W2, slhs);
  k_srhs  <<<dim3((Bq*Nq*Tq + TB-1)/TB), TB, 0, stream>>>(Xt, W3, srhs);
  k_sigP  <<<dim3(((size_t)Bq*Nq*Nq + TB-1)/TB), TB, 0, stream>>>(slhs, srhs, bs, sigPb);
  k_cvt_bf<<<dim3(((size_t)Nq*Nq + TB-1)/TB), TB, 0, stream>>>(Vs, Vsbf, (size_t)Nq*Nq);
  k_gemm_S<<<dim3(Nq/64, Nq/64, Bq), 128, 0, stream>>>(Vsbf, sigPb, S);
  k_softmaxS<<<dim3(Nq/32, Bq), dim3(32, 8), 0, stream>>>(S);
  k_diag  <<<dim3((Bq*Nq + TB-1)/TB), TB, 0, stream>>>(S, dgS);
  // Laplacian edge norms
  k_zero  <<<dim3((Nq + TB-1)/TB), TB, 0, stream>>>(deg, Nq);
  k_deg   <<<dim3((Eq + TB-1)/TB), TB, 0, stream>>>(ei, deg);
  k_dis   <<<dim3((Nq + TB-1)/TB), TB, 0, stream>>>(deg, dis);
  k_norm  <<<dim3((Eq + TB-1)/TB), TB, 0, stream>>>(ei, dis, nrm);
  k_attnorm<<<dim3((Bq*Eq + TB-1)/TB), TB, 0, stream>>>(nrm, S, ei, attn);
  // Chebyshev terms
  const size_t nTx = (size_t)Bq*Mq*Fq;
  k_Tx0   <<<dim3((nTx + TB-1)/TB), TB, 0, stream>>>(X, dgS, Tx0);
  k_cvt_bf<<<dim3((3*Fq*Cq + TB-1)/TB), TB, 0, stream>>>(chW, chWbf, (size_t)3*Fq*Cq);
  k_gemm_cheb<<<dim3(Mq/64, Bq), 128, 0, stream>>>(Tx0, chWbf + 0*Fq*Cq, acc, 0);
  k_zero  <<<dim3((nTx + TB-1)/TB), TB, 0, stream>>>(Tx1, nTx);
  k_prop  <<<dim3(Eq, Bq), 256, 0, stream>>>(Tx1, Tx0, attn, Eq, 1.f, ei);
  k_gemm_cheb<<<dim3(Mq/64, Bq), 128, 0, stream>>>(Tx1, chWbf + 1*Fq*Cq, acc, 1);
  k_negcopy<<<dim3((nTx + TB-1)/TB), TB, 0, stream>>>(Tx0, Tx2, nTx);
  k_prop  <<<dim3(Eq, Bq), 256, 0, stream>>>(Tx2, Tx1, nrm, 0, 2.f, ei);
  k_gemm_cheb<<<dim3(Mq/64, Bq), 128, 0, stream>>>(Tx2, chWbf + 2*Fq*Cq, acc, 1);
  k_biasrelu<<<dim3(((size_t)Bq*Nq*T2q*Cq + TB-1)/TB), TB, 0, stream>>>(acc, chb);
  // temporal conv (im2col GEMM) + fused epilogue
  k_packWt<<<dim3((192*64 + TB-1)/TB), TB, 0, stream>>>(tW, Wt2);
  k_gemm_time<<<dim3(Mq/64, Bq), 128, 0, stream>>>(acc, Wt2, tb, tout);
  k_final <<<dim3(Bq*Nq*Tq), Cq, 0, stream>>>(tout, X, rW, rb, lg, lb, out);
}
